// ChaosStudentLM_28707561406562
// MI455X (gfx1250) — matrix operand order repaired and checked by reading, not compiled, not run on hardware
//
#include <hip/hip_runtime.h>
#include <stdint.h>

// ---------------------------------------------------------------------------
// ChaosStudentLM forward for MI455X (gfx1250, wave32, WMMA).
// Residual stream kept in f32; all GEMMs run bf16 x bf16 -> f32 accumulate via
// v_wmma_f32_16x16x32_bf16 (fp32 WMMA on CDNA5 is only 16x16x4 -> 8x less K
// per issue, so bf16 is the precision the chip rewards). Weights are converted
// to bf16 once per launch; lm_head is transposed to K-major and padded to a
// multiple of 128 so the 4096 x 50257 logits GEMM (half of all FLOPs) uses the
// same contiguous fragment-load fast path and stays L2-resident (103 MB bf16
// < 192 MB L2).
// ---------------------------------------------------------------------------

#define Bb 2
#define Tt 2048
#define Vv 50257
#define VP 50432        // V padded to multiple of 128
#define Dd 1024
#define Ll 8
#define FF 2048         // FFM * D
#define MM (Bb * Tt)    // 4096 rows

typedef __attribute__((ext_vector_type(16))) __bf16         v16bf;
typedef __attribute__((ext_vector_type(8)))  float          v8f;
typedef __attribute__((ext_vector_type(8)))  unsigned short ushort8;
typedef __attribute__((ext_vector_type(16))) unsigned short ushort16;
typedef __attribute__((ext_vector_type(4)))  unsigned short ushort4v;

__device__ __forceinline__ unsigned short f32_bf16(float f) {
  unsigned u = __builtin_bit_cast(unsigned, f);
  u += 0x7FFFu + ((u >> 16) & 1u);            // round-to-nearest-even
  return (unsigned short)(u >> 16);
}

// ---------------- embedding gather: x[row,:] = embed[id[row],:] -------------
__global__ __launch_bounds__(256) void k_embed(const int* __restrict__ ids,
                                               const float* __restrict__ emb,
                                               float* __restrict__ x) {
  const int row = blockIdx.x;
  const int id  = ids[row];
  const float4* src = (const float4*)(emb + (size_t)id * Dd);
  float4* dst = (float4*)(x + (size_t)row * Dd);
  dst[threadIdx.x] = src[threadIdx.x];        // 256 thr * 4 f32 = 1024
}

// ---------------- f32 -> bf16 bulk convert ---------------------------------
__global__ __launch_bounds__(256) void k_f32_to_bf16(const float* __restrict__ in,
                                                     unsigned short* __restrict__ out,
                                                     size_t n) {
  size_t i = (size_t)blockIdx.x * blockDim.x + threadIdx.x;
  if (i < n) out[i] = f32_bf16(in[i]);
}

// ------------- lm_head (V,D) f32 -> (D,VP) bf16, zero-padded ---------------
__global__ __launch_bounds__(256) void k_lmhead_t(const float* __restrict__ lm,
                                                  unsigned short* __restrict__ out) {
  __shared__ unsigned short tile[32][33];
  const int v0 = blockIdx.x * 32;             // along V (padded)
  const int d0 = blockIdx.y * 32;             // along D
#pragma unroll
  for (int k = 0; k < 32; k += 8) {
    const int v = v0 + threadIdx.y + k;
    const float val = (v < Vv) ? lm[(size_t)v * Dd + d0 + threadIdx.x] : 0.f;
    tile[threadIdx.y + k][threadIdx.x] = f32_bf16(val);
  }
  __syncthreads();
#pragma unroll
  for (int k = 0; k < 32; k += 8)
    out[(size_t)(d0 + threadIdx.y + k) * VP + v0 + threadIdx.x] =
        tile[threadIdx.x][threadIdx.y + k];
}

// ---------------- rmsnorm row -> bf16 --------------------------------------
__global__ __launch_bounds__(256) void k_rmsnorm_bf16(const float* __restrict__ x,
                                                      const float* __restrict__ w,
                                                      unsigned short* __restrict__ out) {
  const int row = blockIdx.x;
  const float4 v = ((const float4*)(x + (size_t)row * Dd))[threadIdx.x];
  float ss = v.x * v.x + v.y * v.y + v.z * v.z + v.w * v.w;
#pragma unroll
  for (int off = 16; off; off >>= 1) ss += __shfl_down(ss, off, 32);
  __shared__ float red[8];
  if ((threadIdx.x & 31) == 0) red[threadIdx.x >> 5] = ss;
  __syncthreads();
  float tot = 0.f;
#pragma unroll
  for (int i = 0; i < 8; ++i) tot += red[i];
  const float inv = rsqrtf(tot * (1.f / Dd) + 1e-6f);
  const float4 wv = ((const float4*)w)[threadIdx.x];
  ushort4v o;
  o.x = f32_bf16(v.x * inv * wv.x);
  o.y = f32_bf16(v.y * inv * wv.y);
  o.z = f32_bf16(v.z * inv * wv.z);
  o.w = f32_bf16(v.w * inv * wv.w);
  *(ushort4v*)(out + (size_t)row * Dd + threadIdx.x * 4) = o;
}

// --------------- chaotic scan: s_t = tanh(a*s + pre_t), hs -> bf16 ----------
__global__ __launch_bounds__(256) void k_scan(const float* __restrict__ pre,
                                              const float* __restrict__ a_diag,
                                              unsigned short* __restrict__ hs) {
  const int idx = blockIdx.x * blockDim.x + threadIdx.x;   // 0 .. B*D-1
  const int b = idx >> 10;                                  // / D
  const int d = idx & (Dd - 1);
  const float a = a_diag[d];
  const float* p = pre + (size_t)b * Tt * Dd + d;
  unsigned short* h = hs + (size_t)b * Tt * Dd + d;
  float s = 0.f;
  for (int t = 0; t < Tt; ++t) {
    s = tanhf(a * s + p[(size_t)t * Dd]);
    h[(size_t)t * Dd] = f32_bf16(s);
  }
}

// ---------------- WMMA bf16 GEMM -------------------------------------------
// C[M=4096, N] = A(bf16, MxK row-major) @ B(bf16, K x ldb row-major)
// flags: bit0 = SiLU after bias, bit1 = bf16 output. res != null -> += res.
__global__ __launch_bounds__(256) void k_gemm_bf16(
    const unsigned short* __restrict__ A, const unsigned short* __restrict__ Bw,
    const float* __restrict__ bias, const float* __restrict__ res,
    void* __restrict__ Cout, int K, int ldb, int ldc, int Nvalid, int flags) {
  const int lane = threadIdx.x & 31;
  const int wv   = threadIdx.x >> 5;          // 8 waves / block
  const int wr   = wv & 3;                    // 4 row groups
  const int wc   = wv >> 2;                   // 2 col groups
  const int m0   = blockIdx.y * 128 + wr * 32;
  const int n0   = blockIdx.x * 128 + wc * 64;
  const int arow = m0 + (lane & 15);
  const int koff = (lane >> 4) << 3;          // A layout: lanes 16-31 hold K+8

  const v8f vzero = {0.f, 0.f, 0.f, 0.f, 0.f, 0.f, 0.f, 0.f};
  v8f acc[2][4];
#pragma unroll
  for (int t = 0; t < 2; ++t)
#pragma unroll
    for (int c = 0; c < 4; ++c) acc[t][c] = vzero;

  for (int k0 = 0; k0 < K; k0 += 32) {
    v16bf afrag[2], bfrag[4];
#pragma unroll
    for (int t = 0; t < 2; ++t) {
      // 16-bit A 16x32: VGPR0-3 hold K=koff..koff+7, VGPR4-7 hold koff+16..+23
      const unsigned short* ap = A + (size_t)(arow + t * 16) * K + (k0 + koff);
      ushort8 lo = *(const ushort8*)ap;
      ushort8 hi = *(const ushort8*)(ap + 16);
      ushort16 u = __builtin_shufflevector(lo, hi, 0, 1, 2, 3, 4, 5, 6, 7, 8, 9,
                                           10, 11, 12, 13, 14, 15);
      afrag[t] = __builtin_bit_cast(v16bf, u);
    }
    // 16-bit B 32x16: N = lane&15, elements hold K = koff..koff+7, koff+16..+23
    const unsigned short* bp = Bw + (size_t)(k0 + koff) * ldb + n0 + (lane & 15);
#pragma unroll
    for (int c = 0; c < 4; ++c) {
      ushort16 u;
#pragma unroll
      for (int s = 0; s < 16; ++s)
        u[s] = bp[(size_t)((s & 7) + ((s >> 3) << 4)) * ldb + c * 16];
      bfrag[c] = __builtin_bit_cast(v16bf, u);
    }
    if (k0 + 32 < K) {                         // global_prefetch_b8 next K slab
      __builtin_prefetch(bp + (size_t)32 * ldb, 0, 1);
      __builtin_prefetch(A + (size_t)arow * K + k0 + 32, 0, 1);
    }
#pragma unroll
    for (int t = 0; t < 2; ++t)
#pragma unroll
      for (int c = 0; c < 4; ++c)
        acc[t][c] = __builtin_amdgcn_wmma_f32_16x16x32_bf16(
            false, afrag[t], false, bfrag[c], (short)0, acc[t][c], false, false);
  }

  // C/D layout: lanes 0-15 -> M = vgpr, lanes 16-31 -> M = vgpr + 8; N = lane&15
  const int mhalf = (lane >> 4) << 3;
#pragma unroll
  for (int t = 0; t < 2; ++t) {
    const int mbase = m0 + t * 16 + mhalf;
#pragma unroll
    for (int c = 0; c < 4; ++c) {
      const int ncol = n0 + c * 16 + (lane & 15);
      if (ncol >= Nvalid) continue;
      const float bv = bias ? bias[ncol] : 0.f;
#pragma unroll
      for (int e = 0; e < 8; ++e) {
        const int m = mbase + e;
        float val = acc[t][c][e] + bv;
        if (flags & 1) val = val / (1.f + __expf(-val));    // SiLU
        if (res) val += res[(size_t)m * ldc + ncol];
        if (flags & 2)
          ((unsigned short*)Cout)[(size_t)m * ldc + ncol] = f32_bf16(val);
        else
          ((float*)Cout)[(size_t)m * ldc + ncol] = val;
      }
    }
  }
}

// ---------------------------------------------------------------------------
extern "C" void kernel_launch(void* const* d_in, const int* in_sizes, int n_in,
                              void* d_out, int out_size, void* d_ws, size_t ws_size,
                              hipStream_t stream) {
  (void)in_sizes; (void)n_in; (void)out_size; (void)ws_size;
  const int*   ids   = (const int*)d_in[0];
  const float* emb   = (const float*)d_in[1];
  const float* n1w   = (const float*)d_in[2];
  const float* n2w   = (const float*)d_in[3];
  const float* adiag = (const float*)d_in[4];
  const float* Win   = (const float*)d_in[5];
  const float* bin   = (const float*)d_in[6];
  const float* Wout  = (const float*)d_in[7];
  const float* W1    = (const float*)d_in[8];
  const float* b1    = (const float*)d_in[9];
  const float* W2    = (const float*)d_in[10];
  const float* b2    = (const float*)d_in[11];
  const float* fnw   = (const float*)d_in[12];
  const float* lmw   = (const float*)d_in[13];
  float* logits = (float*)d_out;

  // ---- workspace carve-out (~271 MB) ----
  char* p = (char*)d_ws;
  auto carve = [&](size_t bytes) -> char* {
    char* r = p;
    p += (bytes + 255) & ~(size_t)255;
    return r;
  };
  float*          xf    = (float*)carve((size_t)MM * Dd * 4);
  float*          pre   = (float*)carve((size_t)MM * Dd * 4);
  unsigned short* hbf   = (unsigned short*)carve((size_t)MM * Dd * 2);
  unsigned short* hsbf  = (unsigned short*)carve((size_t)MM * Dd * 2);
  unsigned short* hidbf = (unsigned short*)carve((size_t)MM * FF * 2);
  unsigned short* WinB  = (unsigned short*)carve((size_t)Ll * Dd * Dd * 2);
  unsigned short* WoutB = (unsigned short*)carve((size_t)Ll * Dd * Dd * 2);
  unsigned short* W1B   = (unsigned short*)carve((size_t)Ll * Dd * FF * 2);
  unsigned short* W2B   = (unsigned short*)carve((size_t)Ll * Dd * FF * 2);
  unsigned short* LhT   = (unsigned short*)carve((size_t)Dd * VP * 2);

  const dim3 blk256(256);
  auto cvt = [&](const float* src, unsigned short* dst, size_t n) {
    k_f32_to_bf16<<<dim3((unsigned)((n + 255) / 256)), blk256, 0, stream>>>(src, dst, n);
  };
  auto gemm = [&](const unsigned short* A, const unsigned short* Bm,
                  const float* bias, const float* res, void* C, int K, int N,
                  int ldc, int Nvalid, int flags) {
    k_gemm_bf16<<<dim3(N / 128, MM / 128), blk256, 0, stream>>>(
        A, Bm, bias, res, C, K, N, ldc, Nvalid, flags);
  };

  // ---- one-time per-launch weight conversion (deterministic, no caching) ----
  cvt(Win,  WinB,  (size_t)Ll * Dd * Dd);
  cvt(Wout, WoutB, (size_t)Ll * Dd * Dd);
  cvt(W1,   W1B,   (size_t)Ll * Dd * FF);
  cvt(W2,   W2B,   (size_t)Ll * Dd * FF);
  k_lmhead_t<<<dim3(VP / 32, Dd / 32), dim3(32, 8), 0, stream>>>(lmw, LhT);

  // ---- embedding ----
  k_embed<<<dim3(MM), blk256, 0, stream>>>(ids, emb, xf);

  // ---- layers ----
  for (int l = 0; l < Ll; ++l) {
    // norm1 -> bf16
    k_rmsnorm_bf16<<<dim3(MM), blk256, 0, stream>>>(xf, n1w + (size_t)l * Dd, hbf);
    // pre = h @ W_in + b_in  (f32 out)
    gemm(hbf, WinB + (size_t)l * Dd * Dd, bin + (size_t)l * Dd, nullptr, pre,
         Dd, Dd, Dd, Dd, 0);
    // sequential chaotic scan -> hs (bf16)
    k_scan<<<dim3((Bb * Dd) / 256), blk256, 0, stream>>>(
        pre, adiag + (size_t)l * Dd, hsbf);
    // x += hs @ W_out
    gemm(hsbf, WoutB + (size_t)l * Dd * Dd, nullptr, xf, xf, Dd, Dd, Dd, Dd, 0);
    // norm2 -> bf16
    k_rmsnorm_bf16<<<dim3(MM), blk256, 0, stream>>>(xf, n2w + (size_t)l * Dd, hbf);
    // hid = silu(h @ ff_w1 + b1) -> bf16
    gemm(hbf, W1B + (size_t)l * Dd * FF, b1 + (size_t)l * FF, nullptr, hidbf,
         Dd, FF, FF, FF, 1 | 2);
    // x += hid @ ff_w2 + b2
    gemm(hidbf, W2B + (size_t)l * Dd * FF, b2 + (size_t)l * Dd, xf, xf,
         FF, Dd, Dd, Dd, 0);
  }

  // ---- final norm + logits ----
  k_rmsnorm_bf16<<<dim3(MM), blk256, 0, stream>>>(xf, fnw, hbf);
  // logits (M x V) = hbf @ LhT ; B padded to VP, stores guarded to V, ldc = V
  gemm(hbf, LhT, nullptr, nullptr, logits, Dd, VP, Vv, Vv, 0);
}
